// TropLinear_18794776887633
// MI455X (gfx1250) — matrix operand order repaired and checked by reading, not compiled, not run on hardware
//
#include <hip/hip_runtime.h>

// ---------------------------------------------------------------------------
// TropLinear for MI455X (gfx1250, wave32, WMMA)
// B=4 S=512 IN=1024 OUT=512 ; T=16 G=2 K=4 R=32 ; tokens = 2048
// Strategy: token-dependent cell gather -> masked dense bf16 GEMM.
// Each cell (t,g,k) spans exactly one R=32 slice == one v_wmma_f32_16x16x32_bf16
// K-step, so per-cell A = (winner==k ? z : 0) is a cheap vector select on a
// fragment loaded once per wave. Each wave drives FOUR output tiles with one
// shared A-select (pipelined one cell ahead), so WMMA hazard slots are filled
// by independent WMMAs / loads / next-select cndmasks instead of v_nops.
// ---------------------------------------------------------------------------

typedef __attribute__((ext_vector_type(16))) __bf16 v16bf;
typedef __attribute__((ext_vector_type(8)))  float  v8f;

#define TOKENS 2048
#define IN_DIM 1024
#define OUT_DIM 512
#define RANK 32
#define NTG 32           // T*G
#define NCELL 128        // T*G*K
#define OUT_SCALE 0.17677669529663687f   // 1/sqrt(32)

// ---------------- Kernel 1: projection + routing + winner pack --------------
// grid 256 blocks x 256 thr ; 8 waves/block, one token per wave.
__global__ __launch_bounds__(256) void trop_route(
    const float* __restrict__ x, const float* __restrict__ proj_w,
    const float* __restrict__ router_w, const float* __restrict__ router_b,
    __bf16* __restrict__ zbf, unsigned long long* __restrict__ wpack)
{
    __shared__ float ps[32][65];   // proj chunk, padded vs bank conflicts
    __shared__ float xs[8][64];    // x chunk per block-token
    __shared__ float zl[8][32];    // z exchange per wave
    __shared__ unsigned char wl[8][32];

    const int lane = threadIdx.x & 31;
    const int wave = threadIdx.x >> 5;
    const int tok_base = blockIdx.x * 8;
    const int token = tok_base + wave;

    float acc = 0.0f;
    for (int ch = 0; ch < 16; ++ch) {
        for (int e = threadIdx.x; e < 32 * 64; e += 256) {
            int r = e >> 6, i = e & 63;
            ps[r][i] = proj_w[r * IN_DIM + ch * 64 + i];
        }
        for (int e = threadIdx.x; e < 8 * 64; e += 256) {
            int tb = e >> 6, i = e & 63;
            xs[tb][i] = x[(size_t)(tok_base + tb) * IN_DIM + ch * 64 + i];
        }
        __syncthreads();
        #pragma unroll 8
        for (int i = 0; i < 64; ++i)
            acc += xs[wave][i] * ps[lane][i];
        __syncthreads();
    }
    // lane r of wave holds z[token][r]
    zbf[(size_t)token * RANK + lane] = (__bf16)acc;
    zl[wave][lane] = acc;
    __syncthreads();

    // lane == tg ; compute 4 cell scores, argmin (first-min like jnp.argmin)
    int win = 0;
    float best = 0.0f;
    #pragma unroll
    for (int kk = 0; kk < 4; ++kk) {
        int cell = lane * 4 + kk;
        float s = router_b[cell];
        #pragma unroll 8
        for (int r = 0; r < RANK; ++r)
            s += zl[wave][r] * router_w[cell * RANK + r];
        if (kk == 0) { best = s; win = 0; }
        else if (s < best) { best = s; win = kk; }
    }
    wl[wave][lane] = (unsigned char)win;
    __syncthreads();
    if (lane < 2) {
        unsigned int p = 0;
        #pragma unroll
        for (int t = 0; t < 16; ++t)
            p |= ((unsigned int)(wl[wave][lane * 16 + t] & 3)) << (2 * t);
        ((unsigned int*)wpack)[token * 2 + lane] = p;
    }
}

// ------------- Kernel 2a: affine_w fp32 -> bf16, B-fragment swizzle ---------
// Layout: wswz[cell][out_tile][lane][16]  with element j of lane l being
// W[cell][ r = 16*(l>>4) + j ][ out_tile*16 + (l&15) ]  (contiguous k order, same as A:
// lanes 0-15 hold K=0..15, lanes 16-31 hold K=16..31).
__global__ __launch_bounds__(256) void trop_swizzle_w(
    const float* __restrict__ aw, __bf16* __restrict__ wswz)
{
    size_t idx = (size_t)blockIdx.x * 256 + threadIdx.x;   // 2^21 total
    int j  = (int)(idx & 15);
    int l  = (int)((idx >> 4) & 31);
    int ot = (int)((idx >> 9) & 31);
    int c  = (int)(idx >> 14);
    int r  = ((l >> 4) << 4) + j;
    int o  = ot * 16 + (l & 15);
    wswz[idx] = (__bf16)aw[((size_t)c * RANK + r) * OUT_DIM + o];
}

// ------------- Kernel 2b: per-token winner-gathered bias sum ----------------
__global__ __launch_bounds__(256) void trop_bias(
    const float* __restrict__ ab, const unsigned long long* __restrict__ wpack,
    float* __restrict__ bias_sum)
{
    int idx = blockIdx.x * 256 + threadIdx.x;   // 2048*512
    int o = idx & (OUT_DIM - 1);
    int tok = idx >> 9;
    unsigned long long wp = wpack[tok];
    float s = 0.0f;
    #pragma unroll 8
    for (int tg = 0; tg < NTG; ++tg) {
        int kk = (int)((wp >> (2 * tg)) & 3ull);
        s += ab[(tg * 4 + kk) * OUT_DIM + o];
    }
    bias_sum[idx] = s;
}

// ------------- Kernel 3: masked bf16 WMMA gather-GEMM -----------------------
// grid (128, 1) x 256 thr. Each wave: one 16-token tile x FOUR 16-col out
// tiles (independent accumulators), A-select pipelined one cell ahead.
__global__ __launch_bounds__(256) void trop_main(
    const __bf16* __restrict__ zbf, const unsigned long long* __restrict__ wpack,
    const float* __restrict__ bias_sum, const __bf16* __restrict__ wswz,
    float* __restrict__ out)
{
    const int lane = threadIdx.x & 31;
    const int wave = threadIdx.x >> 5;
    const int tok0 = blockIdx.x * 16;
    const int ot0  = wave * 4;                      // out tiles ot0 .. ot0+3
    const int h    = lane >> 4;
    const int ml   = lane & 15;

    // --- A fragment (loaded once): contiguous k order (same as B), VGPR j: K = 2j + 16h
    const int token_m = tok0 + ml;
    const __bf16* zrow = zbf + (size_t)token_m * RANK;
    v16bf afull;
    #pragma unroll
    for (int j = 0; j < 8; ++j) {
        int k = 2 * j + 16 * h;
        afull[2 * j]     = zrow[k];
        afull[2 * j + 1] = zrow[k + 1];
    }
    v16bf azero;
    #pragma unroll
    for (int e = 0; e < 16; ++e) azero[e] = (__bf16)0.0f;

    const unsigned long long wp = wpack[token_m];
    // base pointers for this wave's four B-fragment streams
    const __bf16* bbase0 = wswz + ((size_t)(ot0 + 0) * 32 + lane) * 16;
    const __bf16* bbase1 = wswz + ((size_t)(ot0 + 1) * 32 + lane) * 16;
    const __bf16* bbase2 = wswz + ((size_t)(ot0 + 2) * 32 + lane) * 16;
    const __bf16* bbase3 = wswz + ((size_t)(ot0 + 3) * 32 + lane) * 16;

    v8f acc0 = {}, acc1 = {}, acc2 = {}, acc3 = {};

    // select for cell 0 (32-bit extraction -> v_cmp_eq_u32)
    v16bf afrag = (((unsigned int)wp & 3u) == 0u) ? afull : azero;

    #pragma unroll 4
    for (int c = 0; c < NCELL; ++c) {
        const size_t coff = (size_t)c * (32 * 32 * 16);
        // eight coalesced b128 loads (pre-swizzled B layout, L2-resident)
        const v16bf bfrag0 = *(const v16bf*)(bbase0 + coff);
        const v16bf bfrag1 = *(const v16bf*)(bbase1 + coff);
        const v16bf bfrag2 = *(const v16bf*)(bbase2 + coff);
        const v16bf bfrag3 = *(const v16bf*)(bbase3 + coff);

        // pipeline: NEXT cell's A-select into a fresh buffer so its cndmasks
        // co-execute in this cell's WMMA hazard slots
        v16bf anext;
        if (c + 1 < NCELL) {
            const int cn = c + 1;
            const unsigned int w2n =
                ((unsigned int)(wp >> (2 * (cn >> 2)))) & 3u;
            anext = (w2n == (unsigned)(cn & 3)) ? afull : azero;
        } else {
            anext = azero;
        }

        acc0 = __builtin_amdgcn_wmma_f32_16x16x32_bf16(
            false, afrag, false, bfrag0, (short)0, acc0, false, false);
        acc1 = __builtin_amdgcn_wmma_f32_16x16x32_bf16(
            false, afrag, false, bfrag1, (short)0, acc1, false, false);
        acc2 = __builtin_amdgcn_wmma_f32_16x16x32_bf16(
            false, afrag, false, bfrag2, (short)0, acc2, false, false);
        acc3 = __builtin_amdgcn_wmma_f32_16x16x32_bf16(
            false, afrag, false, bfrag3, (short)0, acc3, false, false);

        afrag = anext;
    }

    // --- epilogue: C/D layout M = v + 8h, N = lane&15
    #pragma unroll
    for (int v = 0; v < 8; ++v) {
        const int token = tok0 + v + 8 * h;
        const size_t oi = (size_t)token * OUT_DIM + ot0 * 16 + ml;
        out[oi]      = (acc0[v] + bias_sum[oi])      * OUT_SCALE;
        out[oi + 16] = (acc1[v] + bias_sum[oi + 16]) * OUT_SCALE;
        out[oi + 32] = (acc2[v] + bias_sum[oi + 32]) * OUT_SCALE;
        out[oi + 48] = (acc3[v] + bias_sum[oi + 48]) * OUT_SCALE;
    }
}

// ---------------------------------------------------------------------------
extern "C" void kernel_launch(void* const* d_in, const int* in_sizes, int n_in,
                              void* d_out, int out_size, void* d_ws, size_t ws_size,
                              hipStream_t stream)
{
    const float* x        = (const float*)d_in[0];
    const float* proj_w   = (const float*)d_in[1];
    const float* router_w = (const float*)d_in[2];
    const float* router_b = (const float*)d_in[3];
    const float* affine_w = (const float*)d_in[4];
    const float* affine_b = (const float*)d_in[5];
    float* out = (float*)d_out;

    // workspace layout (~8.4 MB)
    char* ws = (char*)d_ws;
    __bf16*             zbf      = (__bf16*)(ws);                       // 2048*32*2   = 128 KB
    unsigned long long* wpack    = (unsigned long long*)(ws + 131072);  // 2048*8      =  16 KB
    float*              bias_sum = (float*)(ws + 147456);               // 2048*512*4  =   4 MB
    __bf16*             wswz     = (__bf16*)(ws + 147456 + 4194304);    // 2^21*2      =   4 MB

    trop_route<<<dim3(TOKENS / 8), dim3(256), 0, stream>>>(
        x, proj_w, router_w, router_b, zbf, wpack);

    trop_swizzle_w<<<dim3((NCELL * 32 * 32 * 16) / 256), dim3(256), 0, stream>>>(
        affine_w, wswz);

    trop_bias<<<dim3((TOKENS * OUT_DIM) / 256), dim3(256), 0, stream>>>(
        affine_b, wpack, bias_sum);

    trop_main<<<dim3(TOKENS / 16, 1), dim3(256), 0, stream>>>(
        zbf, wpack, bias_sum, wswz, out);
}
